// SparseResidualBlock_66383014527054
// MI455X (gfx1250) — matrix operand order repaired and checked by reading, not compiled, not run on hardware
//
#include <hip/hip_runtime.h>
#include <hip/hip_bf16.h>

typedef __attribute__((ext_vector_type(16))) _Float16 v16h;
typedef __attribute__((ext_vector_type(8)))  _Float16 v8h;
typedef __attribute__((ext_vector_type(4)))  _Float16 h4;
typedef __attribute__((ext_vector_type(8)))  float    v8f;

#define CCH 128
#define KOFF 27
#define ROWP 136           // padded LDS row stride (f16): 272 B, multiple of 16 B
#define NBLK 256           // BN partial-reduction blocks

// ---------------------------------------------------------------------------
// BN statistics: per-block partial sums (deterministic two-stage reduction)
// ---------------------------------------------------------------------------
__global__ void bn_partial(const float* __restrict__ X,
                           float* __restrict__ ps, float* __restrict__ ps2,
                           int rows_per_block) {
    int c  = threadIdx.x;                 // 128 threads: one per channel
    int r0 = blockIdx.x * rows_per_block;
    float s = 0.f, s2 = 0.f;
    for (int r = 0; r < rows_per_block; ++r) {
        float v = X[(size_t)(r0 + r) * CCH + c];
        s += v; s2 += v * v;
    }
    ps [blockIdx.x * CCH + c] = s;
    ps2[blockIdx.x * CCH + c] = s2;
}

__global__ void bn_finalize(const float* __restrict__ ps, const float* __restrict__ ps2,
                            const float* __restrict__ gamma, const float* __restrict__ beta,
                            float* __restrict__ scale, float* __restrict__ shift,
                            int nblocks, float invN) {
    int c = threadIdx.x;                  // 128 threads
    float s = 0.f, s2 = 0.f;
    for (int b = 0; b < nblocks; ++b) { s += ps[b * CCH + c]; s2 += ps2[b * CCH + c]; }
    float mu  = s * invN;
    float var = s2 * invN - mu * mu;
    if (var < 0.f) var = 0.f;
    float sc = gamma[c] * rsqrtf(var + 1e-4f);
    scale[c] = sc;
    shift[c] = beta[c] - mu * sc;
}

// BN + ReLU, f32 -> f16, 4 channels per thread
__global__ void bn_relu_f16(const float* __restrict__ X,
                            const float* __restrict__ scale, const float* __restrict__ shift,
                            _Float16* __restrict__ H, int total4) {
    int i = blockIdx.x * blockDim.x + threadIdx.x;
    if (i >= total4) return;
    int c4 = (i & 31) << 2;               // 32 groups of 4 channels per row
    float4 v = ((const float4*)X)[i];
    float a0 = fmaxf(v.x * scale[c4 + 0] + shift[c4 + 0], 0.f);
    float a1 = fmaxf(v.y * scale[c4 + 1] + shift[c4 + 1], 0.f);
    float a2 = fmaxf(v.z * scale[c4 + 2] + shift[c4 + 2], 0.f);
    float a3 = fmaxf(v.w * scale[c4 + 3] + shift[c4 + 3], 0.f);
    h4 o = { (_Float16)a0, (_Float16)a1, (_Float16)a2, (_Float16)a3 };
    ((h4*)H)[i] = o;
}

// ---------------------------------------------------------------------------
// Pack W[k][cin][cout] (f32) into WMMA B-fragment order (f16):
// Wp[k][cc][strip][lane][e], e=0..15; element e of lane maps to
// B row = cc*32 + (lane/16)*16 + e, col = strip*16 + lane%16.
// ---------------------------------------------------------------------------
__global__ void pack_weights(const float* __restrict__ W, _Float16* __restrict__ Wp) {
    int i = blockIdx.x * blockDim.x + threadIdx.x;
    if (i >= (KOFF << 14)) return;        // 27*4*8*32*16
    int e    =  i        & 15;
    int lane = (i >> 4)  & 31;
    int s    = (i >> 9)  & 7;
    int cc   = (i >> 12) & 3;
    int k    =  i >> 14;
    int krow = cc * 32 + ((lane >> 4) << 4) + e;
    int col  = (s << 4) + (lane & 15);
    Wp[i] = (_Float16)W[((size_t)k * CCH + krow) * CCH + col];
}

// ---------------------------------------------------------------------------
// Submanifold conv via WMMA: each workgroup = 64 sites x 128 cout.
// 8 waves; wave w owns cout strip [16w, 16w+16), 4 row tiles of 16.
// Software pipeline: gather k+1 into registers while WMMAs consume LDS
// buffer k&1; drain into buffer (k+1)&1; single barrier per iteration.
// Masking is applied branchlessly (AND with 0/all-ones) AFTER an
// unconditional global_load_b128 so no flat/scratch path is generated.
// ---------------------------------------------------------------------------
__device__ __forceinline__ uint4 and_mask(uint4 a, unsigned m) {
    a.x &= m; a.y &= m; a.z &= m; a.w &= m;
    return a;
}

__global__ __launch_bounds__(256)
void subm_conv_wmma(const _Float16* __restrict__ H,     // [N][128] f16
                    const _Float16* __restrict__ Wp,    // packed fragments
                    const int* __restrict__ nbr_idx,    // [N][27]
                    const int* __restrict__ nbr_mask,   // [N][27]
                    const float* __restrict__ bias,     // [128]
                    const float* __restrict__ resid,    // [N][128] or nullptr
                    float* __restrict__ Y) {            // [N][128]
    __shared__ _Float16 At[2][64 * ROWP];               // 2 x 17408 B
    const int tid   = threadIdx.x;
    const int wave  = tid >> 5;
    const int lane  = tid & 31;
    const int khalf = lane >> 4;
    const int m0    = blockIdx.x * 64;
    const int col   = (wave << 4) + (lane & 15);

    // Each thread stages 2 (row,seg) slots: rows row0, row0+32, same seg.
    const int row0 = tid >> 3;
    const int seg  = tid & 7;                           // 16 f16 = 32 B

    const float bval = bias[col];
    v8f acc[4];
#pragma unroll
    for (int t = 0; t < 4; ++t)
#pragma unroll
        for (int r = 0; r < 8; ++r) acc[t][r] = bval;

    uint4 rlo[2], rhi[2];

    // Prologue: gather k=0 into registers, store to buffer 0.
#pragma unroll
    for (int j = 0; j < 2; ++j) {
        int site = m0 + row0 + 32 * j;
        int g    = nbr_idx [(size_t)site * KOFF + 0];
        unsigned m = nbr_mask[(size_t)site * KOFF + 0] ? 0xFFFFFFFFu : 0u;
        const _Float16* src = H + (size_t)g * CCH + (seg << 4);
        rlo[j] = and_mask(*(const uint4*)(src),     m);   // idx always valid
        rhi[j] = and_mask(*(const uint4*)(src + 8), m);
    }
#pragma unroll
    for (int j = 0; j < 2; ++j) {
        _Float16* dst = &At[0][(row0 + 32 * j) * ROWP + (seg << 4)];
        *(uint4*)(dst)     = rlo[j];
        *(uint4*)(dst + 8) = rhi[j];
    }
    __syncthreads();

    for (int k = 0; k < KOFF; ++k) {
        const _Float16* buf = At[k & 1];
        const bool more = (k + 1) < KOFF;

        // Issue next-k gathers (outstanding while WMMAs run below).
        if (more) {
#pragma unroll
            for (int j = 0; j < 2; ++j) {
                int site = m0 + row0 + 32 * j;
                int g    = nbr_idx [(size_t)site * KOFF + (k + 1)];
                unsigned m = nbr_mask[(size_t)site * KOFF + (k + 1)] ? 0xFFFFFFFFu : 0u;
                const _Float16* src = H + (size_t)g * CCH + (seg << 4);
                rlo[j] = and_mask(*(const uint4*)(src),     m);
                rhi[j] = and_mask(*(const uint4*)(src + 8), m);
            }
        }

        // Compute: 4 cin chunks x 4 m-tiles of WMMA from LDS buffer k&1.
#pragma unroll
        for (int cc = 0; cc < 4; ++cc) {
            v16h bfrag = *(const v16h*)(Wp + ((size_t)(((k * 4 + cc) * 8 + wave) * 32 + lane) << 4));
            const int kcol0 = cc * 32 + (khalf << 4);
#pragma unroll
            for (int t = 0; t < 4; ++t) {
                const _Float16* ap = buf + (t * 16 + (lane & 15)) * ROWP + kcol0;
                v8h lo = *(const v8h*)(ap);
                v8h hi = *(const v8h*)(ap + 8);
                v16h afrag = __builtin_shufflevector(lo, hi,
                    0, 1, 2, 3, 4, 5, 6, 7, 8, 9, 10, 11, 12, 13, 14, 15);
                acc[t] = __builtin_amdgcn_wmma_f32_16x16x32_f16(
                    false, afrag, false, bfrag, (short)0, acc[t], false, false);
            }
        }

        // Drain next-k gathers into the other buffer.
        if (more) {
            _Float16* nbuf = At[(k + 1) & 1];
#pragma unroll
            for (int j = 0; j < 2; ++j) {
                _Float16* dst = nbuf + (row0 + 32 * j) * ROWP + (seg << 4);
                *(uint4*)(dst)     = rlo[j];
                *(uint4*)(dst + 8) = rhi[j];
            }
        }
        __syncthreads();
    }

    // C/D layout: VGPR r -> row = khalf*8 + r within tile, col = lane%16.
#pragma unroll
    for (int t = 0; t < 4; ++t) {
        int rbase = m0 + t * 16 + (khalf << 3);
#pragma unroll
        for (int r = 0; r < 8; ++r) {
            size_t off = (size_t)(rbase + r) * CCH + col;
            float v = acc[t][r];
            if (resid) v += resid[off];
            Y[off] = v;
        }
    }
}

// ---------------------------------------------------------------------------
extern "C" void kernel_launch(void* const* d_in, const int* in_sizes, int n_in,
                              void* d_out, int out_size, void* d_ws, size_t ws_size,
                              hipStream_t stream) {
    const float* features = (const float*)d_in[0];
    const int*   nbr_idx  = (const int*)  d_in[1];
    const int*   nbr_mask = (const int*)  d_in[2];
    const float* gamma1   = (const float*)d_in[3];
    const float* beta1    = (const float*)d_in[4];
    const float* W1       = (const float*)d_in[5];
    const float* bias1    = (const float*)d_in[6];
    const float* gamma2   = (const float*)d_in[7];
    const float* beta2    = (const float*)d_in[8];
    const float* W2       = (const float*)d_in[9];
    const float* bias2    = (const float*)d_in[10];

    const int N = in_sizes[0] / CCH;

    char* base = (char*)d_ws;
    _Float16* H   = (_Float16*)base;                                   // N*128*2
    float*    Y1  = (float*)(base + (size_t)N * CCH * 2);              // N*128*4
    char* p = base + (size_t)N * CCH * 2 + (size_t)N * CCH * 4;
    _Float16* Wp1 = (_Float16*)p;  p += (size_t)KOFF * CCH * CCH * 2;
    _Float16* Wp2 = (_Float16*)p;  p += (size_t)KOFF * CCH * CCH * 2;
    float* ps     = (float*)p;     p += (size_t)NBLK * CCH * 4;
    float* ps2    = (float*)p;     p += (size_t)NBLK * CCH * 4;
    float* scale1 = (float*)p;     p += 512;
    float* shift1 = (float*)p;     p += 512;
    float* scale2 = (float*)p;     p += 512;
    float* shift2 = (float*)p;     p += 512;

    const int rpb    = N / NBLK;
    const int total4 = N * (CCH / 4);
    const float invN = 1.0f / (float)N;
    const int packTot = KOFF << 14;

    // Weights (independent of activations)
    pack_weights<<<(packTot + 255) / 256, 256, 0, stream>>>(W1, Wp1);
    pack_weights<<<(packTot + 255) / 256, 256, 0, stream>>>(W2, Wp2);

    // BN1 + ReLU -> H (f16)
    bn_partial <<<NBLK, CCH, 0, stream>>>(features, ps, ps2, rpb);
    bn_finalize<<<1, CCH, 0, stream>>>(ps, ps2, gamma1, beta1, scale1, shift1, NBLK, invN);
    bn_relu_f16<<<(total4 + 255) / 256, 256, 0, stream>>>(features, scale1, shift1, H, total4);

    // Conv1 -> Y1 (f32)
    subm_conv_wmma<<<N / 64, 256, 0, stream>>>(H, Wp1, nbr_idx, nbr_mask, bias1, nullptr, Y1);

    // BN2 + ReLU -> H (f16, reuse)
    bn_partial <<<NBLK, CCH, 0, stream>>>(Y1, ps, ps2, rpb);
    bn_finalize<<<1, CCH, 0, stream>>>(ps, ps2, gamma2, beta2, scale2, shift2, NBLK, invN);
    bn_relu_f16<<<(total4 + 255) / 256, 256, 0, stream>>>(Y1, scale2, shift2, H, total4);

    // Conv2 + residual -> d_out (f32)
    subm_conv_wmma<<<N / 64, 256, 0, stream>>>(H, Wp2, nbr_idx, nbr_mask, bias2,
                                               features, (float*)d_out);
}
